// SurpriseGatedStore_6794638262894
// MI455X (gfx1250) — matrix operand order repaired and checked by reading, not compiled, not run on hardware
//
#include <hip/hip_runtime.h>
#include <math.h>

typedef __attribute__((ext_vector_type(16))) __bf16 v16bf;
typedef __attribute__((ext_vector_type(8)))  __bf16 v8bf;
typedef __attribute__((ext_vector_type(4)))  __bf16 v4bf;
typedef __attribute__((ext_vector_type(8)))  float  v8f;

#define EPSN 1e-8f

// ---------- bf16 hi/lo split helpers (fp32 ~= hi + lo) ----------
__device__ __forceinline__ unsigned short f2bf_rne(float f) {
  unsigned u = __float_as_uint(f);
  u += 0x7FFFu + ((u >> 16) & 1u);          // round-to-nearest-even
  return (unsigned short)(u >> 16);
}
__device__ __forceinline__ float bf2f(unsigned short h) {
  return __uint_as_float(((unsigned)h) << 16);
}
__device__ __forceinline__ __bf16 bfbits(unsigned short h) {
  return __builtin_bit_cast(__bf16, h);
}

// =====================================================================
// Kernel 0: one-time weight split  W(fp32) -> hi/lo bf16 planes in ws.
// grid 256 blocks x 256 threads, 4 elements/thread (262144 per matrix)
// =====================================================================
__global__ void wconv_kernel(const float* __restrict__ W,
                             __bf16* __restrict__ hi,
                             __bf16* __restrict__ lo) {
  const int i = (blockIdx.x * 256 + threadIdx.x) * 4;
  const float4 v = *reinterpret_cast<const float4*>(W + i);
  float f[4] = {v.x, v.y, v.z, v.w};
  v4bf h, l;
#pragma unroll
  for (int e = 0; e < 4; ++e) {
    unsigned short hb = f2bf_rne(f[e]);
    h[e] = bfbits(hb);
    l[e] = bfbits(f2bf_rne(f[e] - bf2f(hb)));
  }
  *reinterpret_cast<v4bf*>(hi + i) = h;
  *reinterpret_cast<v4bf*>(lo + i) = l;
}

// =====================================================================
// Kernel 1a: partial sums of x over L  (x: [8,8192,512])
// grid 256 blocks (b*32+chunk), 512 threads (one per d)
// =====================================================================
__global__ void pool_partial_kernel(const float* __restrict__ x,
                                    float* __restrict__ partial) {
  const int blk = blockIdx.x;
  const int b = blk >> 5, chunk = blk & 31;
  const int d = threadIdx.x;
  const float* p = x + ((size_t)b * 8192 + (size_t)chunk * 256) * 512 + d;
  float s = 0.0f;
#pragma unroll 4
  for (int l = 0; l < 256; ++l) s += p[(size_t)l * 512];
  partial[(size_t)blk * 512 + d] = s;
}

// =====================================================================
// Kernel 1b: pooled mean + pooled norms.  grid 8 blocks, 512 threads
// =====================================================================
__global__ void pool_final_kernel(const float* __restrict__ partial,
                                  float* __restrict__ pooled,
                                  float* __restrict__ npool) {
  __shared__ float red[512];
  const int b = blockIdx.x, d = threadIdx.x;
  float s = 0.0f;
#pragma unroll
  for (int c = 0; c < 32; ++c) s += partial[((size_t)(b * 32 + c)) * 512 + d];
  s *= (1.0f / 8192.0f);
  pooled[b * 512 + d] = s;
  red[d] = s * s;
  __syncthreads();
  for (int off = 256; off; off >>= 1) {
    if (d < off) red[d] += red[d + off];
    __syncthreads();
  }
  if (d == 0) npool[b] = fmaxf(sqrtf(red[0]), EPSN);
}

// =====================================================================
// Kernel 2: cosine scan over S=131072 embeddings.
// grid 1024 blocks x 256 threads; one wave per embedding, 16 emb/wave.
// Inactive slots are skipped before loading the 2KB row (halves traffic).
// Per-block winners packed as order-preserving (key<<32 | idx) u64.
// =====================================================================
__global__ void __launch_bounds__(256) sim_kernel(
    const float* __restrict__ emb, const float* __restrict__ surprise,
    const float* __restrict__ pooled, const float* __restrict__ npool,
    unsigned long long* __restrict__ blockbest) {
  __shared__ unsigned long long wbest[8][8];
  const int wave = threadIdx.x >> 5, lane = threadIdx.x & 31;

  // Each lane caches its 16-column slice of all 8 pooled vectors in regs.
  float pr[8][16];
  float pnv[8];
#pragma unroll
  for (int b = 0; b < 8; ++b) {
    pnv[b] = npool[b];
#pragma unroll
    for (int j = 0; j < 16; ++j) pr[b][j] = pooled[b * 512 + lane * 16 + j];
  }

  unsigned long long best[8];
#pragma unroll
  for (int b = 0; b < 8; ++b) best[b] = 0ull;

  const int sbase = blockIdx.x * 128 + wave * 16;
  for (int i = 0; i < 16; ++i) {
    const int s = sbase + i;
    if (surprise[s] > 0.0f) {                     // active slot only
      const float* er = emb + (size_t)s * 512 + lane * 16;
      float ev[16];
#pragma unroll
      for (int j = 0; j < 16; ++j) ev[j] = er[j];
      float acc[9];
#pragma unroll
      for (int q = 0; q < 9; ++q) acc[q] = 0.0f;
#pragma unroll
      for (int j = 0; j < 16; ++j) {
        const float e = ev[j];
        acc[8] += e * e;
#pragma unroll
        for (int b = 0; b < 8; ++b) acc[b] += e * pr[b][j];
      }
#pragma unroll
      for (int q = 0; q < 9; ++q) {               // wave32 butterfly reduce
        float v = acc[q];
        v += __shfl_xor(v, 1);  v += __shfl_xor(v, 2);
        v += __shfl_xor(v, 4);  v += __shfl_xor(v, 8);
        v += __shfl_xor(v, 16);
        acc[q] = v;
      }
      const float ne = fmaxf(sqrtf(acc[8]), EPSN);
#pragma unroll
      for (int b = 0; b < 8; ++b) {
        const float sim = acc[b] / (ne * pnv[b]);
        unsigned fb = __float_as_uint(sim);
        unsigned key = (fb & 0x80000000u) ? ~fb : (fb | 0x80000000u);
        unsigned long long pk =
            (((unsigned long long)key) << 32) | (unsigned)s;
        if (pk > best[b]) best[b] = pk;
      }
    }
  }
  if (lane == 0) {
#pragma unroll
    for (int b = 0; b < 8; ++b) wbest[wave][b] = best[b];
  }
  __syncthreads();
  if (threadIdx.x < 8) {
    const int b = threadIdx.x;
    unsigned long long m = 0ull;
#pragma unroll
    for (int w = 0; w < 8; ++w) m = (wbest[w][b] > m) ? wbest[w][b] : m;
    blockbest[(size_t)blockIdx.x * 8 + b] = m;
  }
}

// =====================================================================
// Kernel 3: per-b global argmax, gather best embedding, norm, gate.
// grid 8 blocks x 256 threads
// =====================================================================
__global__ void best_kernel(const unsigned long long* __restrict__ blockbest,
                            const float* __restrict__ emb,
                            float* __restrict__ best_stored,
                            float* __restrict__ g2v, float* __restrict__ nbv,
                            int nblocks) {
  __shared__ unsigned long long red[256];
  __shared__ float fr[256];
  const int b = blockIdx.x, t = threadIdx.x;
  unsigned long long m = 0ull;
  for (int i = t; i < nblocks; i += 256) {
    unsigned long long v = blockbest[(size_t)i * 8 + b];
    if (v > m) m = v;
  }
  red[t] = m;
  __syncthreads();
  for (int off = 128; off; off >>= 1) {
    if (t < off && red[t + off] > red[t]) red[t] = red[t + off];
    __syncthreads();
  }
  const unsigned long long pk = red[0];
  const bool has = (pk != 0ull);
  const unsigned idx = (unsigned)(pk & 0xFFFFFFFFull);
  const unsigned key = (unsigned)(pk >> 32);
  const unsigned fb = (key & 0x80000000u) ? (key ^ 0x80000000u) : ~key;
  const float maxsim = __uint_as_float(fb);

  float nn = 0.0f;
  for (int i = t; i < 512; i += 256) {
    const float v = has ? emb[(size_t)idx * 512 + i] : 0.0f;
    best_stored[b * 512 + i] = v;
    nn += v * v;
  }
  fr[t] = nn;
  __syncthreads();
  for (int off = 128; off; off >>= 1) {
    if (t < off) fr[t] += fr[t + off];
    __syncthreads();
  }
  if (t == 0) {
    nbv[b] = fmaxf(sqrtf(fr[0]), EPSN);
    g2v[b] = (has && maxsim > 0.3f) ? 2.0f : 0.0f;  // pattern_match * 2
  }
}

// =====================================================================
// GEMM stage: A (bf16 hi/lo planes in LDS) x W^T (bf16 hi/lo planes in
// global/L2)  ->  LDS output.  64 rows x 512 cols per block, 8 waves.
// Wave w: row-band (w&3), col-set (w>>2), 4 chunks of 64 cols,
// K-loop 16 x (K=32), 3 x v_wmma per 16x16 tile (hi*hi + lo*hi + hi*lo).
// Register double-buffered: fragments for kt+1 are loaded while the
// WMMAs for kt execute (occupancy is LDS-bound, so VGPRs are free).
// mode: 0 = linear -> bf16 hi/lo planes, 1 = gelu -> bf16 hi/lo planes,
//       2 = linear -> fp32 plane.
// =====================================================================
struct FragSet {
  v16bf ah, al;
  v16bf bh[4], bl[4];
};

__device__ __forceinline__ void mma12(v8f acc[4], const FragSet& f) {
#pragma unroll
  for (int t = 0; t < 4; ++t) {
    acc[t] = __builtin_amdgcn_wmma_f32_16x16x32_bf16(
        false, f.ah, false, f.bh[t], (short)0, acc[t], false, false);
    acc[t] = __builtin_amdgcn_wmma_f32_16x16x32_bf16(
        false, f.al, false, f.bh[t], (short)0, acc[t], false, false);
    acc[t] = __builtin_amdgcn_wmma_f32_16x16x32_bf16(
        false, f.ah, false, f.bl[t], (short)0, acc[t], false, false);
  }
}

__device__ __forceinline__ void gemm_stage(const __bf16* __restrict__ aHi,
                                           const __bf16* __restrict__ aLo,
                                           void* __restrict__ outBuf,
                                           const __bf16* __restrict__ wHi,
                                           const __bf16* __restrict__ wLo,
                                           const float* __restrict__ bias,
                                           int wave, int lane, int mode) {
  const int band = wave & 3;
  const int colset = wave >> 2;
  const int half = lane >> 4;
  const int lan16 = lane & 15;
  const int mrow = band * 16 + lan16;

#pragma unroll 1
  for (int c = 0; c < 4; ++c) {
    const int colBase = colset * 64 + c * 128;
    size_t woff[4];
#pragma unroll
    for (int t = 0; t < 4; ++t)
      woff[t] = (size_t)(colBase + t * 16 + lan16) * 512 + half * 16;

    auto loadF = [&](int kt, FragSet& f) {
      const int k0 = kt * 32;
      // A fragment (16x32 bf16): lane-lo K 0-15, lane-hi K 16-31
      const int ab = mrow * 512 + k0 + half * 16;
      v8bf a0h = *reinterpret_cast<const v8bf*>(aHi + ab);
      v8bf a1h = *reinterpret_cast<const v8bf*>(aHi + ab + 8);
      v8bf a0l = *reinterpret_cast<const v8bf*>(aLo + ab);
      v8bf a1l = *reinterpret_cast<const v8bf*>(aLo + ab + 8);
      f.ah = __builtin_shufflevector(a0h, a1h, 0, 1, 2, 3, 4, 5, 6, 7, 8, 9,
                                     10, 11, 12, 13, 14, 15);
      f.al = __builtin_shufflevector(a0l, a1l, 0, 1, 2, 3, 4, 5, 6, 7, 8, 9,
                                     10, 11, 12, 13, 14, 15);
      // B fragments (32x16 bf16): lane n holds W[n][k0 + half*16 .. +15]
#pragma unroll
      for (int t = 0; t < 4; ++t) {
        f.bh[t] = *reinterpret_cast<const v16bf*>(wHi + woff[t] + k0);
        f.bl[t] = *reinterpret_cast<const v16bf*>(wLo + woff[t] + k0);
      }
    };

    v8f acc[4] = {{}, {}, {}, {}};
    FragSet f0, f1;
    loadF(0, f0);
#pragma unroll 1
    for (int kt = 0; kt < 14; kt += 2) {
      loadF(kt + 1, f1);   // prefetch next K-step before consuming f0
      mma12(acc, f0);
      loadF(kt + 2, f0);   // prefetch kt+2 before consuming f1
      mma12(acc, f1);
    }
    loadF(15, f1);
    mma12(acc, f0);
    mma12(acc, f1);

    // epilogue: bias (+gelu), convert once, scatter C tile
#pragma unroll
    for (int t = 0; t < 4; ++t) {
      const int n = colBase + t * 16 + lan16;
      const float bb = bias[n];
      const int r0 = band * 16 + half * 8;
#pragma unroll
      for (int r = 0; r < 8; ++r) {
        float v = acc[t][r] + bb;
        if (mode == 1) v = 0.5f * v * (1.0f + erff(v * 0.70710678118f));
        const int idx = (r0 + r) * 512 + n;
        if (mode == 2) {
          reinterpret_cast<float*>(outBuf)[idx] = v;
        } else {
          __bf16* oh = reinterpret_cast<__bf16*>(outBuf);
          __bf16* ol = oh + 64 * 512;
          const unsigned short hb = f2bf_rne(v);
          oh[idx] = bfbits(hb);
          ol[idx] = bfbits(f2bf_rne(v - bf2f(hb)));
        }
      }
    }
  }
}

__global__ void __launch_bounds__(256) mlp_surprise_kernel(
    const float* __restrict__ x, const __bf16* __restrict__ WcH,
    const __bf16* __restrict__ WcL, const float* __restrict__ bc,
    const __bf16* __restrict__ W1H, const __bf16* __restrict__ W1L,
    const float* __restrict__ b1, const __bf16* __restrict__ W2H,
    const __bf16* __restrict__ W2L, const float* __restrict__ b2,
    const float* __restrict__ best_stored, const float* __restrict__ g2v,
    const float* __restrict__ nbv, float* __restrict__ out) {
  extern __shared__ __align__(16) char smem_raw[];
  // Region A (128KB): bf16 hi/lo planes (or fp32 scratch after stage 3)
  // Region B (128KB): bf16 hi/lo planes / fp32 pred plane for stage 3
  __bf16* bufA = reinterpret_cast<__bf16*>(smem_raw);
  __bf16* bufB = reinterpret_cast<__bf16*>(smem_raw + 128 * 1024);
  float* predF = reinterpret_cast<float*>(smem_raw + 128 * 1024);
  float* bsF = reinterpret_cast<float*>(smem_raw);

  const int m0 = blockIdx.x * 64;
  const int bidx = m0 >> 13;  // batch index (64 | 8192)
  const int l0 = m0 & 8191;

  // Load causal context rows (x shifted by one along L; l==0 -> zeros),
  // converting fp32 -> bf16 hi/lo planes exactly once per element.
  {
    __bf16* aH = bufA;
    __bf16* aL = bufA + 64 * 512;
    for (int i = threadIdx.x; i < 64 * 128; i += 256) {
      const int r = i >> 7;
      const int c4 = (i & 127) * 4;
      const int l = l0 + r;
      float4 v;
      if (l == 0) v = make_float4(0.f, 0.f, 0.f, 0.f);
      else
        v = *reinterpret_cast<const float4*>(x + (size_t)(m0 + r - 1) * 512 +
                                             c4);
      float f[4] = {v.x, v.y, v.z, v.w};
      v4bf h, lo4;
#pragma unroll
      for (int e = 0; e < 4; ++e) {
        const unsigned short hb = f2bf_rne(f[e]);
        h[e] = bfbits(hb);
        lo4[e] = bfbits(f2bf_rne(f[e] - bf2f(hb)));
      }
      *reinterpret_cast<v4bf*>(aH + r * 512 + c4) = h;
      *reinterpret_cast<v4bf*>(aL + r * 512 + c4) = lo4;
    }
  }
  __syncthreads();

  const int wave = threadIdx.x >> 5, lane = threadIdx.x & 31;
  // ctx = context@Wc^T + bc            (A: bufA -> out: bufB planes)
  gemm_stage(bufA, bufA + 64 * 512, bufB, WcH, WcL, bc, wave, lane, 0);
  __syncthreads();
  // h = gelu(ctx@W1^T + b1)            (A: bufB -> out: bufA planes)
  gemm_stage(bufB, bufB + 64 * 512, bufA, W1H, W1L, b1, wave, lane, 1);
  __syncthreads();
  // pred = h@W2^T + b2                 (A: bufA -> out: bufB fp32 plane)
  gemm_stage(bufA, bufA + 64 * 512, predF, W2H, W2L, b2, wave, lane, 2);
  __syncthreads();

  // Stage this batch's best-stored embedding into (now free) region A
  for (int i = threadIdx.x; i < 512; i += 256)
    bsF[i] = best_stored[bidx * 512 + i];
  __syncthreads();

  // Fused epilogue: mse -> sqrt, ||x_row||, dot(x_row, best), gate, max
  const float g2 = g2v[bidx];
  const float nb = nbv[bidx];
  const int row = threadIdx.x >> 2;  // 4 lanes per row
  const int seg = threadIdx.x & 3;
  const float* xr = x + (size_t)(m0 + row) * 512 + seg * 128;
  const float* pr = predF + row * 512 + seg * 128;
  const float* br = bsF + seg * 128;
  float se = 0.f, xx = 0.f, xd = 0.f;
#pragma unroll 4
  for (int c = 0; c < 128; c += 4) {
    float4 xv = *reinterpret_cast<const float4*>(xr + c);
    float4 pv = *reinterpret_cast<const float4*>(pr + c);
    float4 bv = *reinterpret_cast<const float4*>(br + c);
    float e0 = xv.x - pv.x, e1 = xv.y - pv.y, e2 = xv.z - pv.z,
          e3 = xv.w - pv.w;
    se += e0 * e0 + e1 * e1 + e2 * e2 + e3 * e3;
    xx += xv.x * xv.x + xv.y * xv.y + xv.z * xv.z + xv.w * xv.w;
    xd += xv.x * bv.x + xv.y * bv.y + xv.z * bv.z + xv.w * bv.w;
  }
  se += __shfl_xor(se, 1); se += __shfl_xor(se, 2);
  xx += __shfl_xor(xx, 1); xx += __shfl_xor(xx, 2);
  xd += __shfl_xor(xd, 1); xd += __shfl_xor(xd, 2);
  if (seg == 0) {
    const float ps = sqrtf(se * (1.0f / 512.0f));
    const float nx = fmaxf(sqrtf(xx), EPSN);
    const float cosv = xd / (nx * nb);
    const float contr = g2 * (1.0f - cosv);  // pattern*2*divergence
    out[m0 + row] = fmaxf(ps, contr);
  }
}

// =====================================================================
// Host-side launch
// =====================================================================
extern "C" void kernel_launch(void* const* d_in, const int* in_sizes, int n_in,
                              void* d_out, int out_size, void* d_ws,
                              size_t ws_size, hipStream_t stream) {
  (void)in_sizes; (void)n_in; (void)out_size; (void)ws_size;
  const float* x   = (const float*)d_in[0];   // [8,8192,512]
  const float* Wc  = (const float*)d_in[1];   // [512,512]
  const float* bc  = (const float*)d_in[2];   // [512]
  const float* W1  = (const float*)d_in[3];
  const float* b1  = (const float*)d_in[4];
  const float* W2  = (const float*)d_in[5];
  const float* b2  = (const float*)d_in[6];
  const float* emb = (const float*)d_in[7];   // [131072,512]
  const float* sur = (const float*)d_in[8];   // [131072]

  char* ws = (char*)d_ws;
  size_t off = 0;
  float* partial = (float*)(ws + off); off += (size_t)256 * 512 * 4;  // 512KB
  float* pooled  = (float*)(ws + off); off += (size_t)8 * 512 * 4;
  float* npool   = (float*)(ws + off); off += 8 * 4;
  off = (off + 63) & ~(size_t)63;
  unsigned long long* blockbest = (unsigned long long*)(ws + off);
  off += (size_t)1024 * 8 * 8;                                        // 64KB
  float* best_stored = (float*)(ws + off); off += (size_t)8 * 512 * 4;
  float* g2v = (float*)(ws + off); off += 8 * 4;
  float* nbv = (float*)(ws + off); off += 8 * 4;
  off = (off + 63) & ~(size_t)63;
  const size_t WSZ = (size_t)512 * 512;        // elements per weight plane
  __bf16* WcH = (__bf16*)(ws + off); off += WSZ * 2;
  __bf16* WcL = (__bf16*)(ws + off); off += WSZ * 2;
  __bf16* W1H = (__bf16*)(ws + off); off += WSZ * 2;
  __bf16* W1L = (__bf16*)(ws + off); off += WSZ * 2;
  __bf16* W2H = (__bf16*)(ws + off); off += WSZ * 2;
  __bf16* W2L = (__bf16*)(ws + off); off += WSZ * 2;   // +3MB total

  // One-time weight hi/lo split (L2-resident afterwards)
  wconv_kernel<<<256, 256, 0, stream>>>(Wc, WcH, WcL);
  wconv_kernel<<<256, 256, 0, stream>>>(W1, W1H, W1L);
  wconv_kernel<<<256, 256, 0, stream>>>(W2, W2H, W2L);

  pool_partial_kernel<<<256, 512, 0, stream>>>(x, partial);
  pool_final_kernel<<<8, 512, 0, stream>>>(partial, pooled, npool);
  sim_kernel<<<1024, 256, 0, stream>>>(emb, sur, pooled, npool, blockbest);
  best_kernel<<<8, 256, 0, stream>>>(blockbest, emb, best_stored, g2v, nbv,
                                     1024);
  const size_t shmem = (size_t)256 * 1024;  // 2 x 128KB LDS ping-pong
  mlp_surprise_kernel<<<1024, 256, shmem, stream>>>(
      x, WcH, WcL, bc, W1H, W1L, b1, W2H, W2L, b2, best_stored, g2v, nbv,
      (float*)d_out);
}
